// DiTBlock_28054726378305
// MI455X (gfx1250) — matrix operand order repaired and checked by reading, not compiled, not run on hardware
//
#include <hip/hip_runtime.h>
#include <hip/hip_bf16.h>
#include <cstdint>

// ---------------------------------------------------------------------------
// DiT block forward for MI455X (gfx1250, wave32, WMMA 16x16x32 f16).
// All large GEMMs + both attentions run on v_wmma_f32_16x16x32_f16 with f32
// accumulation. Residual stream lives in d_out (fp32). Requires ~115 MiB ws.
// GEMM: block = 8 waves x (32x64 macro-tile) sharing one 64-col weight tile,
// staged to LDS with global_load_async_to_lds_b128 (double-buffered,
// s_wait_asynccnt), so the weight stream is fetched once per block.
// ---------------------------------------------------------------------------

typedef __attribute__((ext_vector_type(16))) _Float16 v16h;
typedef __attribute__((ext_vector_type(8)))  float    v8f;

union Frag {
  v16h h;
  uint4 u[2];
};

constexpr int B_   = 8;
constexpr int N_   = 1024;
constexpr int M_   = 128;
constexpr int C_   = 1024;
constexpr int H_   = 16;
constexpr int Dh_  = 64;
constexpr int MLP_ = 4096;

static __device__ __forceinline__ v8f wmma_f16(const Frag& a, const Frag& b, v8f c) {
  return __builtin_amdgcn_wmma_f32_16x16x32_f16(false, a.h, false, b.h,
                                                (short)0, c, false, false);
}

// CDNA5 async global->LDS copy (16B per lane) + ASYNCcnt wait.
static __device__ __forceinline__ void async_ld_lds16(void* lds, const void* g) {
  uint32_t l = (uint32_t)(uintptr_t)lds;   // low 32 bits = LDS byte offset
  uint64_t a = (uint64_t)(uintptr_t)g;
  asm volatile("global_load_async_to_lds_b128 %0, %1, off" :: "v"(l), "v"(a) : "memory");
}
static __device__ __forceinline__ void wait_async0() {
  asm volatile("s_wait_asynccnt 0" ::: "memory");
}

// ---------------------------------------------------------------------------
// Elementwise helpers
// ---------------------------------------------------------------------------
__global__ void copy_f32x4(const float4* __restrict__ in, float4* __restrict__ out, int n4) {
  int i = blockIdx.x * 256 + threadIdx.x;
  if (i < n4) out[i] = in[i];
}

__global__ void cvt_f32_f16(const float* __restrict__ in, _Float16* __restrict__ out, int n) {
  int i = blockIdx.x * 256 + threadIdx.x;
  if (i < n) out[i] = (_Float16)in[i];
}

// ---------------------------------------------------------------------------
// ada = silu(c_dino) @ W_ada + b_ada   (B x 6C). Tiny: VALU kernel.
// ---------------------------------------------------------------------------
__global__ __launch_bounds__(256)
void ada_kernel(const float* __restrict__ c_dino, const float* __restrict__ W,
                const float* __restrict__ bias, float* __restrict__ ada) {
  __shared__ float ssil[C_];
  const int b = blockIdx.y;
  for (int i = threadIdx.x; i < C_; i += 256) {
    float c = c_dino[b * C_ + i];
    ssil[i] = c / (1.0f + __expf(-c));
  }
  __syncthreads();
  const int j = blockIdx.x * 256 + threadIdx.x;
  float acc = bias[j];
  for (int k = 0; k < C_; ++k) acc += ssil[k] * W[(size_t)k * (6 * C_) + j];
  ada[(size_t)b * (6 * C_) + j] = acc;
}

// ---------------------------------------------------------------------------
// fp32 weight (K x Nc) -> f16 transposed (Nc x K). grid (Nc/32, K/32).
// ---------------------------------------------------------------------------
__global__ __launch_bounds__(256)
void w_to_f16T(const float* __restrict__ W, _Float16* __restrict__ Wt, int K, int Nc) {
  __shared__ _Float16 tile[32][33];
  const int n0 = blockIdx.x * 32, k0 = blockIdx.y * 32;
  const int tx = threadIdx.x & 31, ty = threadIdx.x >> 5;
#pragma unroll
  for (int j = 0; j < 4; ++j)
    tile[ty + j * 8][tx] = (_Float16)W[(size_t)(k0 + ty + j * 8) * Nc + n0 + tx];
  __syncthreads();
#pragma unroll
  for (int j = 0; j < 4; ++j)
    Wt[(size_t)(n0 + ty + j * 8) * K + k0 + tx] = tile[tx][ty + j * 8];
}

// ---------------------------------------------------------------------------
// LayerNorm (eps=1e-6) + adaLN modulate + f32->f16. One block per row.
// ---------------------------------------------------------------------------
__global__ __launch_bounds__(256)
void ln_mod(const float* __restrict__ x, const float* __restrict__ ada,
            _Float16* __restrict__ hout, int seg) {
  const int row = blockIdx.x;
  const int b = row >> 10;
  const float4 v = ((const float4*)(x + (size_t)row * C_))[threadIdx.x];
  float s  = v.x + v.y + v.z + v.w;
  float ss = v.x * v.x + v.y * v.y + v.z * v.z + v.w * v.w;
#pragma unroll
  for (int sh = 1; sh < 32; sh <<= 1) { s += __shfl_xor(s, sh); ss += __shfl_xor(ss, sh); }
  __shared__ float sb[8], ssb[8];
  if ((threadIdx.x & 31) == 0) { sb[threadIdx.x >> 5] = s; ssb[threadIdx.x >> 5] = ss; }
  __syncthreads();
  float st = 0.f, sst = 0.f;
#pragma unroll
  for (int i = 0; i < 8; ++i) { st += sb[i]; sst += ssb[i]; }
  const float mu   = st * (1.0f / C_);
  const float var  = sst * (1.0f / C_) - mu * mu;
  const float rstd = rsqrtf(var + 1e-6f);
  const float* shv = ada + (size_t)b * (6 * C_) + seg * C_;
  const float* scv = shv + C_;
  const int c0 = threadIdx.x * 4;
  _Float16* hp = hout + (size_t)row * C_ + c0;
  const float xs[4] = {v.x, v.y, v.z, v.w};
#pragma unroll
  for (int i = 0; i < 4; ++i) {
    float hn = (xs[i] - mu) * rstd;
    hp[i] = (_Float16)(hn * (1.0f + scv[c0 + i]) + shv[c0 + i]);
  }
}

// ---------------------------------------------------------------------------
// Block-cooperative WMMA GEMM: out = A(f16, rows x K) @ Wt(f16, Nc x K)^T + b.
// Block: 8 waves = 8 row macro-tiles (32 rows each) x one 64-col tile.
// Weight tile (64x32 f16/k-step) async-copied to LDS once per block,
// double-buffered; each wave reads B-fragments with ds_load_b128.
// ---------------------------------------------------------------------------
enum { EPI_QKV = 0, EPI_Q = 1, EPI_KV = 2, EPI_RES = 3, EPI_GELU = 4 };

constexpr int WROW = 40;  // padded row stride (halves): 80B -> conflict-free

template <int EPI>
__global__ __launch_bounds__(256)
void gemm_wmma(const _Float16* __restrict__ A, const _Float16* __restrict__ Wt,
               const float* __restrict__ bias, int K, int Nc,
               float* __restrict__ resOut, _Float16* __restrict__ out16,
               _Float16* __restrict__ qOut, _Float16* __restrict__ kOut,
               _Float16* __restrict__ vTOut) {
  __shared__ __align__(16) _Float16 wtile[2][64 * WROW];
  const int lane = threadIdx.x & 31;
  const int w    = threadIdx.x >> 5;
  const int ctiles = Nc >> 6;
  const int rg  = blockIdx.x / ctiles;   // 256-row group
  const int ctb = blockIdx.x % ctiles;   // 64-col tile
  const int lo = lane & 15, hi = lane >> 4;
  const int rowBase = rg * 256 + w * 32;

  const _Float16* aRow0 = A + (size_t)(rowBase + lo) * K + hi * 8;
  const _Float16* aRow1 = aRow0 + (size_t)16 * K;

  // cooperative weight staging: thread -> 16B of the 64x32 tile
  const int wrow = threadIdx.x >> 2;     // 0..63 (output column within tile)
  const int wseg = threadIdx.x & 3;      // 0..3  (16B segment of 64B row)
  const _Float16* wsrc = Wt + (size_t)(ctb * 64 + wrow) * K + wseg * 8;
  _Float16* wdst[2] = {&wtile[0][wrow * WROW + wseg * 8],
                       &wtile[1][wrow * WROW + wseg * 8]};

  v8f acc[2][4];
#pragma unroll
  for (int t = 0; t < 2; ++t)
#pragma unroll
    for (int i = 0; i < 4; ++i)
#pragma unroll
      for (int r = 0; r < 8; ++r) acc[t][i][r] = 0.0f;

  async_ld_lds16(wdst[0], wsrc);   // stage k-step 0
  wait_async0();
  __syncthreads();

  int ib = 0;
  for (int kk = 0; kk < K; kk += 32, ib ^= 1) {
    if (kk + 32 < K) async_ld_lds16(wdst[ib ^ 1], wsrc + kk + 32);
    if (kk + 256 < K) {  // global_prefetch_b8 ahead of the A stream
      __builtin_prefetch(aRow0 + kk + 256, 0, 1);
      __builtin_prefetch(aRow1 + kk + 256, 0, 1);
    }
    Frag a0, a1;
    a0.u[0] = *(const uint4*)(aRow0 + kk);
    a0.u[1] = *(const uint4*)(aRow0 + kk + 16);
    a1.u[0] = *(const uint4*)(aRow1 + kk);
    a1.u[1] = *(const uint4*)(aRow1 + kk + 16);
    const _Float16* lbase = &wtile[ib][0];
#pragma unroll
    for (int ct = 0; ct < 4; ++ct) {
      const _Float16* lw = lbase + (ct * 16 + lo) * WROW + hi * 8;
      Frag b;
      b.u[0] = ((const uint4*)lw)[0];
      b.u[1] = ((const uint4*)lw)[2];
      acc[0][ct] = wmma_f16(a0, b, acc[0][ct]);
      acc[1][ct] = wmma_f16(a1, b, acc[1][ct]);
    }
    wait_async0();      // next tile resident before anyone reads it
    __syncthreads();    // and this tile fully read before it is overwritten
  }

#pragma unroll
  for (int t = 0; t < 2; ++t) {
#pragma unroll
    for (int ct = 0; ct < 4; ++ct) {
      const int col = ctb * 64 + ct * 16 + lo;
      const float bb = bias[col];
#pragma unroll
      for (int r = 0; r < 8; ++r) {
        const int row = rowBase + t * 16 + r + 8 * hi;
        const float v = acc[t][ct][r] + bb;
        if constexpr (EPI == EPI_RES) {
          resOut[(size_t)row * Nc + col] += v;   // residual (fp32), 1 writer/elem
        } else if constexpr (EPI == EPI_GELU) {
          float g = 0.5f * v * (1.0f + tanhf(0.7978845608f * (v + 0.044715f * v * v * v)));
          out16[(size_t)row * Nc + col] = (_Float16)g;
        } else if constexpr (EPI == EPI_QKV) {
          const int b = row >> 10, n = row & 1023;
          const int sel = col >> 10, c = col & 1023, h = c >> 6, d = c & 63;
          const size_t bh = (size_t)(b * H_ + h);
          if (sel == 0)      qOut[(bh * N_ + n) * Dh_ + d] = (_Float16)v;
          else if (sel == 1) kOut[(bh * N_ + n) * Dh_ + d] = (_Float16)v;
          else               vTOut[(bh * Dh_ + d) * N_ + n] = (_Float16)v;
        } else if constexpr (EPI == EPI_Q) {
          const int b = row >> 10, n = row & 1023;
          const int h = col >> 6, d = col & 63;
          qOut[(((size_t)(b * H_ + h)) * N_ + n) * Dh_ + d] = (_Float16)v;
        } else if constexpr (EPI == EPI_KV) {
          const int b = row >> 7, m = row & 127;
          const int sel = col >> 10, c = col & 1023, h = c >> 6, d = c & 63;
          const size_t bh = (size_t)(b * H_ + h);
          if (sel == 0) kOut[(bh * M_ + m) * Dh_ + d] = (_Float16)v;
          else          vTOut[(bh * Dh_ + d) * M_ + m] = (_Float16)v;
        }
      }
    }
  }
}

// ---------------------------------------------------------------------------
// Flash attention, one wave per (b, h, 16-row q tile). Streams 32 keys/step:
// two 16x16 score tiles (2 WMMAs each, K=Dh=64), online softmax in C-layout,
// P reshaped f32(C-layout) -> f16(A-layout) through 1KB LDS, then 4 WMMAs
// accumulate O over D=64. mask==nullptr -> self-attention.
// ---------------------------------------------------------------------------
__global__ __launch_bounds__(256)
void attn_wmma(const _Float16* __restrict__ q, const _Float16* __restrict__ k,
               const _Float16* __restrict__ vT, const int* __restrict__ mask,
               int Mk, _Float16* __restrict__ out) {
  __shared__ __align__(16) _Float16 plds[8][16 * 32];
  const int lane  = threadIdx.x & 31;
  const int w     = threadIdx.x >> 5;
  const int gwave = (blockIdx.x * 256 + threadIdx.x) >> 5;
  const int nt = gwave & (N_ / 16 - 1);
  const int h  = (gwave >> 6) & (H_ - 1);
  const int b  = gwave >> 10;
  const int lo = lane & 15, hi = lane >> 4;

  const size_t bh = (size_t)(b * H_ + h);
  const _Float16* qp = q + (bh * N_ + nt * 16 + lo) * Dh_ + hi * 16;
  Frag aq0, aq1;
  aq0.u[0] = ((const uint4*)qp)[0];
  aq0.u[1] = *(const uint4*)(qp + 8);
  aq1.u[0] = *(const uint4*)(qp + 32);
  aq1.u[1] = *(const uint4*)(qp + 40);

  const _Float16* kb_ = k  + bh * (size_t)Mk * Dh_;
  const _Float16* vb_ = vT + bh * (size_t)Dh_ * Mk;
  const int* mrow = mask ? (mask + b * Mk) : nullptr;

  float mrun[8], lrun[8];
  v8f o[4];
#pragma unroll
  for (int r = 0; r < 8; ++r) { mrun[r] = -1e30f; lrun[r] = 0.0f; }
#pragma unroll
  for (int i = 0; i < 4; ++i)
#pragma unroll
    for (int r = 0; r < 8; ++r) o[i][r] = 0.0f;

  for (int mt = 0; mt < Mk; mt += 32) {
    v8f s0, s1;
#pragma unroll
    for (int r = 0; r < 8; ++r) { s0[r] = 0.0f; s1[r] = 0.0f; }
    {
      const _Float16* kp = kb_ + (size_t)(mt + lo) * Dh_ + hi * 16;
      Frag bk;
      bk.u[0] = ((const uint4*)kp)[0];
      bk.u[1] = ((const uint4*)kp)[1];
      s0 = wmma_f16(aq0, bk, s0);
      bk.u[0] = ((const uint4*)(kp + 32))[0];
      bk.u[1] = ((const uint4*)(kp + 32))[1];
      s0 = wmma_f16(aq1, bk, s0);
      kp += 16 * Dh_;
      bk.u[0] = ((const uint4*)kp)[0];
      bk.u[1] = ((const uint4*)kp)[1];
      s1 = wmma_f16(aq0, bk, s1);
      bk.u[0] = ((const uint4*)(kp + 32))[0];
      bk.u[1] = ((const uint4*)(kp + 32))[1];
      s1 = wmma_f16(aq1, bk, s1);
    }
    const float sc = 0.125f;  // D^-0.5
    bool m0ok = true, m1ok = true;
    if (mrow) { m0ok = mrow[mt + lo] != 0; m1ok = mrow[mt + 16 + lo] != 0; }
#pragma unroll
    for (int r = 0; r < 8; ++r) {
      float a0 = m0ok ? s0[r] * sc : -1e30f;
      float a1 = m1ok ? s1[r] * sc : -1e30f;
      float amax = fmaxf(a0, a1);
#pragma unroll
      for (int sh = 1; sh < 16; sh <<= 1) amax = fmaxf(amax, __shfl_xor(amax, sh));
      const float mnew  = fmaxf(mrun[r], amax);
      const float alpha = __expf(mrun[r] - mnew);
      const float p0 = __expf(a0 - mnew);
      const float p1 = __expf(a1 - mnew);
      float ps = p0 + p1;
#pragma unroll
      for (int sh = 1; sh < 16; sh <<= 1) ps += __shfl_xor(ps, sh);
      lrun[r] = lrun[r] * alpha + ps;
      mrun[r] = mnew;
      o[0][r] *= alpha; o[1][r] *= alpha; o[2][r] *= alpha; o[3][r] *= alpha;
      const int prow = r + 8 * hi;
      plds[w][prow * 32 + lo]      = (_Float16)p0;
      plds[w][prow * 32 + 16 + lo] = (_Float16)p1;
    }
    Frag pa;  // reload P in A-fragment layout (wave-local LDS transpose)
    pa.u[0] = *(const uint4*)&plds[w][lo * 32 + hi * 16];
    pa.u[1] = *(const uint4*)&plds[w][lo * 32 + hi * 16 + 8];
#pragma unroll
    for (int dt = 0; dt < 4; ++dt) {
      const _Float16* vp = vb_ + (size_t)(dt * 16 + lo) * Mk + mt + hi * 16;
      Frag bv;
      bv.u[0] = ((const uint4*)vp)[0];
      bv.u[1] = ((const uint4*)vp)[1];
      o[dt] = wmma_f16(pa, bv, o[dt]);
    }
  }
#pragma unroll
  for (int r = 0; r < 8; ++r) {
    const float inv = 1.0f / lrun[r];
    const int n = nt * 16 + r + 8 * hi;
    _Float16* op = out + ((size_t)(b * N_ + n)) * C_ + h * Dh_;
#pragma unroll
    for (int dt = 0; dt < 4; ++dt) op[dt * 16 + lo] = (_Float16)(o[dt][r] * inv);
  }
}

// ---------------------------------------------------------------------------
// Host: orchestrate (all launches on `stream`, graph-capture safe)
// ---------------------------------------------------------------------------
extern "C" void kernel_launch(void* const* d_in, const int* in_sizes, int n_in,
                              void* d_out, int out_size, void* d_ws, size_t ws_size,
                              hipStream_t stream) {
  const float* x       = (const float*)d_in[0];
  const float* c_dino  = (const float*)d_in[1];
  const float* c_text  = (const float*)d_in[2];
  const int*   tmask   = (const int*)d_in[3];
  const float* W_ada   = (const float*)d_in[4];
  const float* b_ada   = (const float*)d_in[5];
  const float* W_qkv   = (const float*)d_in[6];
  const float* b_qkv   = (const float*)d_in[7];
  const float* W_psa   = (const float*)d_in[8];
  const float* b_psa   = (const float*)d_in[9];
  const float* W_q     = (const float*)d_in[10];
  const float* b_q     = (const float*)d_in[11];
  const float* W_kv    = (const float*)d_in[12];
  const float* b_kv    = (const float*)d_in[13];
  const float* W_pca   = (const float*)d_in[14];
  const float* b_pca   = (const float*)d_in[15];
  const float* W_fc1   = (const float*)d_in[16];
  const float* b_fc1   = (const float*)d_in[17];
  const float* W_fc2   = (const float*)d_in[18];
  const float* b_fc2   = (const float*)d_in[19];
  (void)in_sizes; (void)n_in; (void)out_size; (void)ws_size;

  constexpr size_t MiB = 1ull << 20;
  char* ws = (char*)d_ws;
  _Float16* h16    = (_Float16*)(ws + 0 * MiB);    // 16 MiB
  float*    adaBuf = (float*)   (ws + 16 * MiB);   // 0.19 MiB
  _Float16* WqkvT  = (_Float16*)(ws + 17 * MiB);   // 6 MiB
  _Float16* WsaT   = (_Float16*)(ws + 23 * MiB);   // 2 MiB
  _Float16* WqT    = (_Float16*)(ws + 25 * MiB);   // 2 MiB
  _Float16* WkvT   = (_Float16*)(ws + 27 * MiB);   // 4 MiB
  _Float16* WcaT   = (_Float16*)(ws + 31 * MiB);   // 2 MiB
  _Float16* Wfc1T  = (_Float16*)(ws + 33 * MiB);   // 8 MiB
  _Float16* Wfc2T  = (_Float16*)(ws + 41 * MiB);   // 8 MiB
  _Float16* ct16   = (_Float16*)(ws + 49 * MiB);   // 2 MiB
  // phase-shared scratch (64 MiB @ 51 MiB)
  _Float16* qb     = (_Float16*)(ws + 51 * MiB);   // SA: 16 MiB | CA q2 | MLP hidden
  _Float16* kb     = (_Float16*)(ws + 67 * MiB);   // SA: 16 MiB | CA k2 (2 MiB)
  _Float16* v2Tb   = (_Float16*)(ws + 69 * MiB);   // CA vT (2 MiB)
  _Float16* vTb    = (_Float16*)(ws + 83 * MiB);   // SA: 16 MiB
  _Float16* attnb  = (_Float16*)(ws + 99 * MiB);   // 16 MiB
  _Float16* mlph   = qb;                           // 64 MiB (reuses SA region)

  float* xcur = (float*)d_out;  // residual stream lives in the output buffer

  // blocks: 256 rows x 64 cols per block
  auto gblocks = [](int rows, int Nc) { return (rows >> 8) * (Nc >> 6); };

  // residual stream init + adaLN vector
  copy_f32x4<<<8192, 256, 0, stream>>>((const float4*)x, (float4*)xcur, (B_ * N_ * C_) / 4);
  ada_kernel<<<dim3(24, 8), 256, 0, stream>>>(c_dino, W_ada, b_ada, adaBuf);

  // weights -> f16 transposed
  w_to_f16T<<<dim3(3 * C_ / 32, C_ / 32), 256, 0, stream>>>(W_qkv, WqkvT, C_, 3 * C_);
  w_to_f16T<<<dim3(C_ / 32, C_ / 32), 256, 0, stream>>>(W_psa, WsaT, C_, C_);
  w_to_f16T<<<dim3(C_ / 32, C_ / 32), 256, 0, stream>>>(W_q, WqT, C_, C_);
  w_to_f16T<<<dim3(2 * C_ / 32, C_ / 32), 256, 0, stream>>>(W_kv, WkvT, C_, 2 * C_);
  w_to_f16T<<<dim3(C_ / 32, C_ / 32), 256, 0, stream>>>(W_pca, WcaT, C_, C_);
  w_to_f16T<<<dim3(MLP_ / 32, C_ / 32), 256, 0, stream>>>(W_fc1, Wfc1T, C_, MLP_);
  w_to_f16T<<<dim3(C_ / 32, MLP_ / 32), 256, 0, stream>>>(W_fc2, Wfc2T, MLP_, C_);
  cvt_f32_f16<<<(B_ * M_ * C_) / 256, 256, 0, stream>>>(c_text, ct16, B_ * M_ * C_);

  // ---- self-attention ----
  ln_mod<<<B_ * N_, 256, 0, stream>>>(xcur, adaBuf, h16, 0);
  gemm_wmma<EPI_QKV><<<gblocks(B_ * N_, 3 * C_), 256, 0, stream>>>(
      h16, WqkvT, b_qkv, C_, 3 * C_, nullptr, nullptr, qb, kb, vTb);
  attn_wmma<<<(B_ * H_ * (N_ / 16)) / 8, 256, 0, stream>>>(qb, kb, vTb, nullptr, N_, attnb);
  gemm_wmma<EPI_RES><<<gblocks(B_ * N_, C_), 256, 0, stream>>>(
      attnb, WsaT, b_psa, C_, C_, xcur, nullptr, nullptr, nullptr, nullptr);

  // ---- cross-attention ----
  ln_mod<<<B_ * N_, 256, 0, stream>>>(xcur, adaBuf, h16, 2);
  gemm_wmma<EPI_Q><<<gblocks(B_ * N_, C_), 256, 0, stream>>>(
      h16, WqT, b_q, C_, C_, nullptr, nullptr, qb, nullptr, nullptr);
  gemm_wmma<EPI_KV><<<gblocks(B_ * M_, 2 * C_), 256, 0, stream>>>(
      ct16, WkvT, b_kv, C_, 2 * C_, nullptr, nullptr, nullptr, kb, v2Tb);
  attn_wmma<<<(B_ * H_ * (N_ / 16)) / 8, 256, 0, stream>>>(qb, kb, v2Tb, tmask, M_, attnb);
  gemm_wmma<EPI_RES><<<gblocks(B_ * N_, C_), 256, 0, stream>>>(
      attnb, WcaT, b_pca, C_, C_, xcur, nullptr, nullptr, nullptr, nullptr);

  // ---- MLP ----
  ln_mod<<<B_ * N_, 256, 0, stream>>>(xcur, adaBuf, h16, 4);
  gemm_wmma<EPI_GELU><<<gblocks(B_ * N_, MLP_), 256, 0, stream>>>(
      h16, Wfc1T, b_fc1, C_, MLP_, nullptr, mlph, nullptr, nullptr, nullptr);
  gemm_wmma<EPI_RES><<<gblocks(B_ * N_, C_), 256, 0, stream>>>(
      mlph, Wfc2T, b_fc2, MLP_, C_, xcur, nullptr, nullptr, nullptr, nullptr);
}
